// MPNNP_44220983279938
// MI455X (gfx1250) — matrix operand order repaired and checked by reading, not compiled, not run on hardware
//
#include <hip/hip_runtime.h>
#include <hip/hip_bf16.h>

#define N_NODES 100000
#define N_EDGES 1600000
#define HDIM    128
#define NLAYERS 3

typedef __attribute__((ext_vector_type(16))) _Float16 v16h;
typedef __attribute__((ext_vector_type(8)))  float    v8f;

// ---------------------------------------------------------------------------
// WMMA 16x16x32 f16 layouts used by this code (wave32; contiguous k, not the ISA table order):
//  A (16x32, MxK): lane L = half*16 + M (half = L>>4). Element e holds
//    K = chunk*32 + kk with kk = half*16 + e.
//    Inverse (M,K)->slot: chunk=K>>5; kk=K&31; half=kk>>4;
//                         e=kk&15; lane=(half<<4)|M.
//  B (32x16, KxN): lane L -> N = L%16, K = (L>>4)*16 + e, e contiguous 0..15.
//  C/D (16x16 f32): lane L, reg r -> M = r + (L>>4)*8, N = L%16.
//
// LDS staging is "fragment-ready": f16 halves stored at
//    ldsA[(chunk*32 + lane)*16 + e]
// so each wave's fragment load is one 32B-aligned v16h read per lane
// (2 x ds_load_b128), with the f32->f16 conversion done once per block.
// ---------------------------------------------------------------------------

__device__ __forceinline__ void stage_frag_pair(_Float16* __restrict__ lds,
                                                int M, int K, float2 f) {
    const int chunk = K >> 5;
    const int kk    = K & 31;
    const int half  = kk >> 4;
    const int e     = kk & 15;
    const int lane  = (half << 4) | M;
    union { _Float16 h[2]; unsigned int u; } pk;
    pk.h[0] = (_Float16)f.x;
    pk.h[1] = (_Float16)f.y;   // K and K+1 land in (e, e+1) of the same slot
    *(unsigned int*)(lds + (chunk * 32 + lane) * 16 + e) = pk.u;
}

__device__ __forceinline__ v16h load_a_frag(const _Float16* __restrict__ lds,
                                            int chunk, int lane) {
    return *(const v16h*)(lds + (chunk * 32 + lane) * 16);
}

// Weights stored K-contiguous per output column j (pre-transposed in prep pass),
// 32B-aligned -> single v16h (2 x b128) load.
__device__ __forceinline__ v16h load_b_frag(const _Float16* __restrict__ B,
                                            int j, int k0, int half) {
    return *(const v16h*)(B + (size_t)j * HDIM + k0 + half * 16);
}

__device__ __forceinline__ float sigmoidf_fast(float v) {
    return 1.0f / (1.0f + __expf(-v));
}

// ---------------------------------------------------------------------------
// One-time prep: f16-convert weights with layouts the B-fragment loader wants.
//   WlT[l][j][k] = weight[l][k][j]   (message GEMM:  m = x @ W_l)
//   Wih[j][k]    = w_ih[j][k]        (GRU: gi = agg @ w_ih^T => B[k,j]=w_ih[j,k])
//   Whh[j][k]    = w_hh[j][k]
// ---------------------------------------------------------------------------
__global__ void mpnn_prep_weights(const float* __restrict__ weight,
                                  const float* __restrict__ w_ih,
                                  const float* __restrict__ w_hh,
                                  _Float16* __restrict__ WlT,
                                  _Float16* __restrict__ Wih,
                                  _Float16* __restrict__ Whh) {
    int i = blockIdx.x * 256 + threadIdx.x;
    if (i < NLAYERS * HDIM * HDIM) {
        int l = i >> 14, rem = i & 16383, j = rem >> 7, k = rem & 127;
        WlT[i] = (_Float16)weight[(l << 14) + (k << 7) + j];
    }
    if (i < 3 * HDIM * HDIM) {
        Wih[i] = (_Float16)w_ih[i];
        Whh[i] = (_Float16)w_hh[i];
    }
}

__global__ void mpnn_zero(float4* __restrict__ p) {
    p[blockIdx.x * 256 + threadIdx.x] = make_float4(0.f, 0.f, 0.f, 0.f);
}

// ---------------------------------------------------------------------------
// m = x @ W_l via WMMA. Block = 32-row strip (two 16-row sub-strips); each of
// the 8 waves owns one 16-col tile and two accumulators, reusing every B
// fragment across both sub-strips.
// ---------------------------------------------------------------------------
__global__ void mpnn_gemm_msg(const float* __restrict__ x,
                              const _Float16* __restrict__ WlT,
                              float* __restrict__ m) {
    __shared__ _Float16 ldsA[2][4 * 32 * 16];   // two fragment-ready 16x128 tiles
    const int r0 = blockIdx.x * 32;

    // Stage 32x128 f32 -> f16 fragment layout. 2048 K-pairs, 8 per thread.
    for (int p = threadIdx.x; p < 2048; p += 256) {
        const int s = p >> 10;                  // sub-strip 0/1
        const int q = p & 1023;
        const int M = q >> 6;
        const int K = (q & 63) << 1;
        float2 f = *(const float2*)(x + (size_t)(r0 + s * 16 + M) * HDIM + K);
        stage_frag_pair(ldsA[s], M, K, f);
    }
    __syncthreads();

    const int lane = threadIdx.x & 31, warp = threadIdx.x >> 5;
    const int c0 = warp * 16, row = lane & 15, half = lane >> 4;

    v8f acc0 = {}, acc1 = {};
#pragma unroll
    for (int t = 0; t < 4; ++t) {
        v16h b  = load_b_frag(WlT, c0 + row, t * 32, half);
        v16h a0 = load_a_frag(ldsA[0], t, lane);
        v16h a1 = load_a_frag(ldsA[1], t, lane);
        acc0 = __builtin_amdgcn_wmma_f32_16x16x32_f16(false, a0, false, b,
                                                      (short)0, acc0, false, false);
        acc1 = __builtin_amdgcn_wmma_f32_16x16x32_f16(false, a1, false, b,
                                                      (short)0, acc1, false, false);
    }
#pragma unroll
    for (int r = 0; r < 8; ++r) {
        m[(size_t)(r0 + r + half * 8) * HDIM + c0 + row]      = acc0[r];
        m[(size_t)(r0 + 16 + r + half * 8) * HDIM + c0 + row] = acc1[r];
    }
}

// ---------------------------------------------------------------------------
// Edge phase: agg[dst] += m[src] * ew. One wave per edge, float4 per lane
// (512B coalesced gather), native f32 atomics; m+agg = 102MB stays L2-resident.
// ---------------------------------------------------------------------------
__global__ void mpnn_edge_scatter(const long long* __restrict__ ei,
                                  const float* __restrict__ ew,
                                  const float* __restrict__ m,
                                  float* __restrict__ agg) {
    const int e = blockIdx.x * 8 + (threadIdx.x >> 5);
    const int lane = threadIdx.x & 31;
    const int src = (int)ei[e];
    const int dst = (int)ei[(size_t)N_EDGES + e];
    const float w = ew[e];
    const float4 v = *(const float4*)(m + (size_t)src * HDIM + lane * 4);
    float* dp = agg + (size_t)dst * HDIM + lane * 4;
    unsafeAtomicAdd(dp + 0, v.x * w);
    unsafeAtomicAdd(dp + 1, v.y * w);
    unsafeAtomicAdd(dp + 2, v.z * w);
    unsafeAtomicAdd(dp + 3, v.w * w);
}

// ---------------------------------------------------------------------------
// Fused GRU cell: gi = agg@w_ih^T + b_ih ; gh = h@w_hh^T + b_hh ; gate math.
// Block = 16-row strip; each of the 8 waves owns a 16-col output chunk and
// computes its 6 accumulator tiles (r/z/n x input/hidden), K=128 each.
// h_old is read f32 from global (L2 hit) to keep the zg*h passthrough exact.
// ---------------------------------------------------------------------------
__global__ void mpnn_gru(const float* __restrict__ agg,
                         const float* __restrict__ xin,
                         const _Float16* __restrict__ Wih,
                         const _Float16* __restrict__ Whh,
                         const float* __restrict__ b_ih,
                         const float* __restrict__ b_hh,
                         float* __restrict__ xout) {
    __shared__ _Float16 ldsT[2][4 * 32 * 16];   // [0]=agg frags, [1]=x frags
    const int r0 = blockIdx.x * 16;

    for (int p = threadIdx.x; p < 2048; p += 256) {
        const int which = p >> 10;
        const int q = p & 1023;
        const int M = q >> 6;
        const int K = (q & 63) << 1;
        const float* srcp = which ? xin : agg;
        float2 f = *(const float2*)(srcp + (size_t)(r0 + M) * HDIM + K);
        stage_frag_pair(ldsT[which], M, K, f);
    }
    __syncthreads();

    const int lane = threadIdx.x & 31, warp = threadIdx.x >> 5;
    const int c0 = warp * 16, row = lane & 15, half = lane >> 4;

    v16h aA[4], aH[4];
#pragma unroll
    for (int t = 0; t < 4; ++t) {
        aA[t] = load_a_frag(ldsT[0], t, lane);
        aH[t] = load_a_frag(ldsT[1], t, lane);
    }

    v8f acc_i[3], acc_h[3];
#pragma unroll
    for (int g = 0; g < 3; ++g) {
        v8f ai = {}, ah = {};
        const int j = g * HDIM + c0 + row;
#pragma unroll
        for (int t = 0; t < 4; ++t) {
            v16h bi = load_b_frag(Wih, j, t * 32, half);
            ai = __builtin_amdgcn_wmma_f32_16x16x32_f16(false, aA[t], false, bi,
                                                        (short)0, ai, false, false);
        }
#pragma unroll
        for (int t = 0; t < 4; ++t) {
            v16h bh = load_b_frag(Whh, j, t * 32, half);
            ah = __builtin_amdgcn_wmma_f32_16x16x32_f16(false, aH[t], false, bh,
                                                        (short)0, ah, false, false);
        }
        acc_i[g] = ai;
        acc_h[g] = ah;
    }

    const int col = c0 + row;
    const float bir = b_ih[col], biz = b_ih[HDIM + col], bin_ = b_ih[2 * HDIM + col];
    const float bhr = b_hh[col], bhz = b_hh[HDIM + col], bhn = b_hh[2 * HDIM + col];

#pragma unroll
    for (int r = 0; r < 8; ++r) {
        const int rr = r + half * 8;
        const float h  = xin[(size_t)(r0 + rr) * HDIM + col];
        const float rg = sigmoidf_fast(acc_i[0][r] + bir + acc_h[0][r] + bhr);
        const float zg = sigmoidf_fast(acc_i[1][r] + biz + acc_h[1][r] + bhz);
        const float ng = tanhf(acc_i[2][r] + bin_ + rg * (acc_h[2][r] + bhn));
        xout[(size_t)(r0 + rr) * HDIM + col] = (1.0f - zg) * ng + zg * h;
    }
}

// ---------------------------------------------------------------------------
extern "C" void kernel_launch(void* const* d_in, const int* in_sizes, int n_in,
                              void* d_out, int out_size, void* d_ws, size_t ws_size,
                              hipStream_t stream) {
    const float*     z      = (const float*)d_in[0];
    const float*     ew     = (const float*)d_in[1];
    const float*     weight = (const float*)d_in[2];
    const float*     w_ih   = (const float*)d_in[3];
    const float*     w_hh   = (const float*)d_in[4];
    const float*     b_ih   = (const float*)d_in[5];
    const float*     b_hh   = (const float*)d_in[6];
    const long long* ei     = (const long long*)d_in[7];
    float*           out    = (float*)d_out;

    const size_t NH = (size_t)N_NODES * HDIM;       // 12.8M floats = 51.2 MB
    float* B0  = (float*)d_ws;
    float* B1  = B0 + NH;
    float* AGG = B1 + NH;
    _Float16* WlT = (_Float16*)(AGG + NH);
    _Float16* Wih = WlT + (size_t)NLAYERS * HDIM * HDIM;
    _Float16* Whh = Wih + (size_t)3 * HDIM * HDIM;

    mpnn_prep_weights<<<192, 256, 0, stream>>>(weight, w_ih, w_hh, WlT, Wih, Whh);

    const int gemm_blocks = N_NODES / 32;           // 3125
    const int gru_blocks  = N_NODES / 16;           // 6250
    const float* xc = z;
    float* m_buf[NLAYERS]  = {B0, B0, B1};
    float* xn_buf[NLAYERS] = {B1, B0, out};

    for (int l = 0; l < NLAYERS; ++l) {
        float* m  = m_buf[l];
        float* xn = xn_buf[l];
        mpnn_gemm_msg<<<gemm_blocks, 256, 0, stream>>>(xc, WlT + (size_t)l * HDIM * HDIM, m);
        mpnn_zero<<<(int)(NH / 4 / 256), 256, 0, stream>>>((float4*)AGG);
        mpnn_edge_scatter<<<N_EDGES / 8, 256, 0, stream>>>(ei, ew, m, AGG);
        mpnn_gru<<<gru_blocks, 256, 0, stream>>>(AGG, xc, Wih, Whh, b_ih, b_hh, xn);
        xc = xn;
    }
}
